// E29cDiagonalElmanCell_53695681134874
// MI455X (gfx1250) — compile-verified, matrix operand order checked by reading, not run on hardware
//
#include <hip/hip_runtime.h>
#include <hip/hip_bf16.h>
#include <math.h>

// Problem constants (from reference): B=8, T=1024, D_IN=512, D=512, N=64
#define BB   8
#define TT   1024
#define DD   512
#define NN   64
#define EE   1024   // 2*D, output width of input projection
#define HD   256    // DD/2: float2 channel pairs per row

typedef __attribute__((ext_vector_type(16))) _Float16 v16h;
typedef __attribute__((ext_vector_type(8)))  float    v8f;
typedef __attribute__((ext_vector_type(2)))  float    f2;   // packed f32 pair -> v_pk_fma_f32
typedef __attribute__((ext_vector_type(4)))  float    f4;   // b128 load granule

// ---------------------------------------------------------------------------
// Kernel 0: pack W_h and W_write (512x512 f32) 2x2 for the recurrent GEMVs:
//   Wpk[k2*HD + p] = { W[2p][2k2], W[2p+1][2k2], W[2p][2k2+1], W[2p+1][2k2+1] }
// so one global_load_b128 feeds a thread's channel pair for TWO K values.
// One-shot 2MB job; coalesced f4 stores, L2-cached gathers.
// ---------------------------------------------------------------------------
__global__ void pack_w(const float* __restrict__ Wh,
                       const float* __restrict__ Ww,
                       f4* __restrict__ Whp,
                       f4* __restrict__ Wwp) {
  const float* src = blockIdx.z ? Ww : Wh;
  f4*          dst = blockIdx.z ? Wwp : Whp;
  const int k2 = blockIdx.y;       // 0..255: K pair index
  const int p  = threadIdx.x;      // 0..255: channel pair index
  const f2 r0 = *(const f2*)(src + (size_t)(2 * p)     * DD + 2 * k2);
  const f2 r1 = *(const f2*)(src + (size_t)(2 * p + 1) * DD + 2 * k2);
  const f4 o = {r0[0], r1[0], r0[1], r1[1]};
  dst[(size_t)k2 * HD + p] = o;
}

// ---------------------------------------------------------------------------
// Kernel 1: xz = x @ W_xz^T   (8192x512)*(512x1024) via v_wmma_f32_16x16x32_f16
// One wave32 owns one 16x16 output tile; K swept in chunks of 32.
// A-fragment (16-bit A 16x32): lanes 0-15 row M=lane, K=0..7 (v0..3) and
// K=16..23 (v4..7); lanes 16-31 row M=lane-16, K=8..15 and 24..31.
// B mirrors it with N in place of M (B fed column-major = rows of W_xz).
// C/D: VGPR r holds row r (lanes 0-15) / row 8+r (lanes 16-31), N = lane%16.
// ---------------------------------------------------------------------------
__global__ void proj_gemm(const float* __restrict__ x,    // (B*T, 512)
                          const float* __restrict__ Wxz,  // (1024, 512)
                          float* __restrict__ xz) {       // (B*T, 1024)
  const int wave  = (blockIdx.x * blockDim.x + threadIdx.x) >> 5;
  const int lane  = threadIdx.x & 31;
  const int ntile = wave & 63;          // 1024/16 = 64 tiles in N
  const int mtile = wave >> 6;          // 8192/16 = 512 tiles in M
  const int mrow  = lane & 15;
  const int half  = lane >> 4;          // 0: lanes 0-15, 1: lanes 16-31

  const int m = mtile * 16 + mrow;      // A source row (same rows both halves)
  const int n = ntile * 16 + mrow;      // B source row of W_xz (= output col)

  v8f c = {};
#pragma unroll 2
  for (int kb = 0; kb < DD; kb += 32) {
    const float* ap = x   + (size_t)m * DD + kb + half * 8;
    const float* bp = Wxz + (size_t)n * DD + kb + half * 8;
    v16h a, b;
#pragma unroll
    for (int i = 0; i < 8; ++i) {
      a[i]     = (_Float16)ap[i];
      a[i + 8] = (_Float16)ap[16 + i];
      b[i]     = (_Float16)bp[i];
      b[i + 8] = (_Float16)bp[16 + i];
    }
    c = __builtin_amdgcn_wmma_f32_16x16x32_f16(
        /*neg_a=*/false, a, /*neg_b=*/false, b,
        /*c_mod=*/(short)0, c, /*reuse_a=*/false, /*reuse_b=*/false);
  }
  const int col = ntile * 16 + mrow;
#pragma unroll
  for (int r = 0; r < 8; ++r) {
    const int row = mtile * 16 + half * 8 + r;
    xz[(size_t)row * EE + col] = c[r];
  }
}

// ---------------------------------------------------------------------------
// Kernel 2: the recurrence. One workgroup per batch, 256 threads (8 wave32s).
// Thread t owns the float2 channel pair (2t, 2t+1). GEMVs read 2x2-packed
// weights with b128 loads (2 K per load) and f4 LDS broadcasts of h (4 K per
// DS op); all math runs on the packed-f32 VALU path (v_pk_fma_f32).
// The 64x512 tape lives in LDS (128KB; CDNA5 has 320KB/WGP).
// ---------------------------------------------------------------------------
__launch_bounds__(256, 1)
__global__ void recurrent(const float* __restrict__ xz,          // (B*T, 1024)
                          const float* __restrict__ h_tape_init, // (B, N, D)
                          const float* __restrict__ h_work_init, // (B, D)
                          const f4* __restrict__ Whp,            // packed W_h
                          const f4* __restrict__ Wwp,            // packed W_write
                          const float* __restrict__ b_h,
                          const float* __restrict__ g_z,
                          const float* __restrict__ g_r,
                          const float* __restrict__ g_h,
                          const float* __restrict__ b_gate,
                          float* __restrict__ outs,              // (B, T, D)
                          float* __restrict__ tape_out) {        // (B, N, D)
  extern __shared__ float smem[];
  float* tape = smem;                 // NN*DD  (f2 view: NN*HD)
  float* hw   = tape + NN * DD;       // DD
  float* hn   = hw + DD;              // DD
  float* wv   = hn + DD;              // DD
  float* sc   = wv + DD;              // NN
  float* eu   = sc + NN;              // NN
  float* attn = eu + NN;              // NN

  f2* tape2 = (f2*)tape;              // [n*HD + p], p = channel pair
  f2* hw2   = (f2*)hw;
  f2* hn2   = (f2*)hn;
  f2* wv2   = (f2*)wv;

  const int b   = blockIdx.x;
  const int tid = threadIdx.x;        // 0..255 == channel pair p
  const float scale = 0.044194173824159216f;  // 1/sqrt(512)

  // ---- init persistent state ----
  const f2* tape_init2 = (const f2*)(h_tape_init + (size_t)b * NN * DD);
  for (int i = tid; i < NN * HD; i += HD) tape2[i] = tape_init2[i];
  hw2[tid] = ((const f2*)(h_work_init + (size_t)b * DD))[tid];
  const f2 bh = ((const f2*)b_h)[tid];
  const f2 gz = ((const f2*)g_z)[tid];
  const f2 gr = ((const f2*)g_r)[tid];
  const f2 gh = ((const f2*)g_h)[tid];
  const f2 bg = ((const f2*)b_gate)[tid];
  __syncthreads();

  const int n_slot = tid >> 2;        // tape slot this thread scores (4 thr/slot)
  const int sub    = tid & 3;         // position within the 4-thread group
  const int kbase  = sub * 64;        // 64 f2-wide K chunk (=128 channels)

  for (int t = 0; t < TT; ++t) {
    const float* xrow = xz + ((size_t)b * TT + t) * EE;
    __builtin_prefetch(xrow + EE, 0, 0);   // global_prefetch: next timestep row

    // ---- read scores: sc[n] = (h_work . tape[n]) * scale ----
    {
      f2 p2 = {};
      const f2* tp = tape2 + n_slot * HD;
#pragma unroll 8
      for (int jj = 0; jj < 64; ++jj) {
        const int j = kbase + ((jj + tid) & 63);   // lane-rotated: conflict-free
        p2 += hw2[j] * tp[j];
      }
      float p = p2[0] + p2[1];
      p += __shfl_down(p, 2, 4);
      p += __shfl_down(p, 1, 4);
      if (sub == 0) sc[n_slot] = p * scale;
    }
    __syncthreads();

    // ---- softmax over 64 slots (redundantly by 64 threads) ----
    if (tid < NN) {
      float m = -1e30f;
      for (int j = 0; j < NN; ++j) m = fmaxf(m, sc[j]);
      eu[tid] = __expf(sc[tid] - m);
    }
    __syncthreads();
    if (tid < NN) {
      float s = 0.f;
      for (int j = 0; j < NN; ++j) s += eu[j];
      attn[tid] = eu[tid] / s;
    }
    __syncthreads();

    // ---- read_val + h_work@W_h^T + tanh (packed channel pair) ----
    f2 rv = {};
#pragma unroll 8
    for (int n = 0; n < NN; ++n) rv += attn[n] * tape2[n * HD + tid];
    f2 m1 = {};
    {
      const f4* wp  = Whp + tid;
      const f4* hwv = (const f4*)hw;     // f4 broadcast: 4 K per DS op
#pragma unroll 4
      for (int k4 = 0; k4 < 128; ++k4) {
        const f4 hk = hwv[k4];
        const f4 w0 = wp[(size_t)(2 * k4)     * HD];
        const f4 w1 = wp[(size_t)(2 * k4 + 1) * HD];
        const f2 a0 = {w0[0], w0[1]};
        const f2 a1 = {w0[2], w0[3]};
        const f2 a2 = {w1[0], w1[1]};
        const f2 a3 = {w1[2], w1[3]};
        m1 += hk[0] * a0;
        m1 += hk[1] * a1;
        m1 += hk[2] * a2;
        m1 += hk[3] * a3;
      }
    }
    const f2 xv = ((const f2*)xrow)[tid];
    f2 pre = xv + m1 + rv + bh;
    f2 h;
    h[0] = tanhf(pre[0]);
    h[1] = tanhf(pre[1]);
    hn2[tid] = h;
    __syncthreads();

    // ---- write_val = h_new @ W_write^T ----
    {
      f2 m2 = {};
      const f4* wp  = Wwp + tid;
      const f4* hnv = (const f4*)hn;
#pragma unroll 4
      for (int k4 = 0; k4 < 128; ++k4) {
        const f4 hk = hnv[k4];
        const f4 w0 = wp[(size_t)(2 * k4)     * HD];
        const f4 w1 = wp[(size_t)(2 * k4 + 1) * HD];
        const f2 a0 = {w0[0], w0[1]};
        const f2 a1 = {w0[2], w0[3]};
        const f2 a2 = {w1[0], w1[1]};
        const f2 a3 = {w1[2], w1[3]};
        m2 += hk[0] * a0;
        m2 += hk[1] * a1;
        m2 += hk[2] * a2;
        m2 += hk[3] * a3;
      }
      wv2[tid] = m2;
    }
    __syncthreads();

    // ---- write scores ----
    {
      f2 p2 = {};
      const f2* tp = tape2 + n_slot * HD;
#pragma unroll 8
      for (int jj = 0; jj < 64; ++jj) {
        const int j = kbase + ((jj + tid) & 63);
        p2 += wv2[j] * tp[j];
      }
      float p = p2[0] + p2[1];
      p += __shfl_down(p, 2, 4);
      p += __shfl_down(p, 1, 4);
      if (sub == 0) sc[n_slot] = p * scale;
    }
    __syncthreads();
    if (tid < NN) {
      float m = -1e30f;
      for (int j = 0; j < NN; ++j) m = fmaxf(m, sc[j]);
      eu[tid] = __expf(sc[tid] - m);
    }
    __syncthreads();
    if (tid < NN) {
      float s = 0.f;
      for (int j = 0; j < NN; ++j) s += eu[j];
      attn[tid] = eu[tid] / s;
    }
    __syncthreads();

    // ---- tape update, gated output, h_work update ----
    {
      const f2 w = wv2[tid];
#pragma unroll 8
      for (int n = 0; n < NN; ++n) {
        const float a = attn[n];
        const f2 tv = tape2[n * HD + tid];
        tape2[n * HD + tid] = tv * (1.f - a) + w * a;
      }
      const f2 zt = ((const f2*)(xrow + DD))[tid];
      f2 gin = zt * gz + rv * gr + h * gh + bg;
      gin[0] = fminf(fmaxf(gin[0], -20.f), 20.f);
      gin[1] = fminf(fmaxf(gin[1], -20.f), 20.f);
      f2 silu;
      silu[0] = gin[0] / (1.f + __expf(-gin[0]));
      silu[1] = gin[1] / (1.f + __expf(-gin[1]));
      ((f2*)(outs + ((size_t)b * TT + t) * DD))[tid] = h * silu;
      hw2[tid] = h;                   // next step's h_work
    }
    __syncthreads();
  }

  // ---- emit final tape ----
  f2* tout2 = (f2*)(tape_out + (size_t)b * NN * DD);
  for (int i = tid; i < NN * HD; i += HD) tout2[i] = tape2[i];
}

// ---------------------------------------------------------------------------
extern "C" void kernel_launch(void* const* d_in, const int* in_sizes, int n_in,
                              void* d_out, int out_size, void* d_ws, size_t ws_size,
                              hipStream_t stream) {
  const float* x           = (const float*)d_in[0];   // (8,1024,512)
  const float* h_tape_init = (const float*)d_in[1];   // (8,64,512)
  const float* h_work_init = (const float*)d_in[2];   // (8,512)
  const float* W_h         = (const float*)d_in[3];   // (512,512)
  const float* W_xz        = (const float*)d_in[4];   // (1024,512)
  const float* b_h         = (const float*)d_in[5];   // (512)
  const float* W_write     = (const float*)d_in[6];   // (512,512)
  const float* g_z         = (const float*)d_in[7];
  const float* g_r         = (const float*)d_in[8];
  const float* g_h         = (const float*)d_in[9];
  const float* b_gate      = (const float*)d_in[10];

  float* ws  = (float*)d_ws;
  float* xz  = ws;                                  // 8192*1024 f32 = 32MB
  f4*    Whp = (f4*)(xz + (size_t)BB * TT * EE);    // 512*512 f32 (packed 2x2)
  f4*    Wwp = Whp + (size_t)DD * DD / 4;           // 512*512 f32 (packed 2x2)

  float* outs     = (float*)d_out;                  // (8,1024,512)
  float* tape_out = outs + (size_t)BB * TT * DD;    // (8,64,512)

  // 0) 2x2-pack weights so recurrent GEMV loads are b128 (2 K per load)
  pack_w<<<dim3(1, HD, 2), HD, 0, stream>>>(W_h, W_write, Whp, Wwp);

  // 1) input projection GEMM on the WMMA path:
  //    512 M-tiles * 64 N-tiles = 32768 wave-tiles, 8 waves / 256-thread block
  proj_gemm<<<4096, 256, 0, stream>>>(x, W_xz, xz);

  // 2) sequential scan: one workgroup per batch, tape resident in LDS
  const size_t smem = (size_t)(NN * DD + 3 * DD + 3 * NN) * sizeof(float); // ~135KB
  recurrent<<<BB, 256, smem, stream>>>(xz, h_tape_init, h_work_init, Whp, Wwp,
                                       b_h, g_z, g_r, g_h, b_gate, outs, tape_out);
}
